// LocalMHA_53772990546330
// MI455X (gfx1250) — matrix operand order repaired and checked by reading, not compiled, not run on hardware
//
#include <hip/hip_runtime.h>

// ---------------------------------------------------------------------------
// LocalMHA on MI455X (gfx1250): f16 WMMA path for QKV / attention / out-proj.
// Wave32, V_WMMA_F32_16X16X32_F16.  GEMM tile staging uses the CDNA5 async
// global->LDS path (ASYNCcnt, double-buffered) when the builtins exist,
// falling back to the register-staged path otherwise.
// ---------------------------------------------------------------------------

typedef _Float16 h16;
typedef __attribute__((ext_vector_type(16))) _Float16 v16h;
typedef __attribute__((ext_vector_type(8)))  float    v8f;

#define DIM     1024
#define T_LEN   16384
#define WINDOW  32
#define DHEAD   64
#define HEADS   16
#define NWIN    (T_LEN / WINDOW)   // 512
#define QKV_N   (3 * DIM)          // 3072

#if defined(__has_builtin)
#  if __has_builtin(__builtin_amdgcn_global_load_async_to_lds_b128) && \
      __has_builtin(__builtin_amdgcn_s_wait_asynccnt)
#    define USE_ASYNC_LDS 1
#  endif
#endif
#ifndef USE_ASYNC_LDS
#  define USE_ASYNC_LDS 0
#endif

// ------------------------- workspace layout (bytes) ------------------------
static const size_t OFF_WQKV = 0;                                    // 3072*1024*2
static const size_t OFF_WOUT = OFF_WQKV + (size_t)QKV_N * DIM * 2;   // 1024*1024*2
static const size_t OFF_NORM = OFF_WOUT + (size_t)DIM * DIM * 2;     // T*1024*2
static const size_t OFF_QKV  = OFF_NORM + (size_t)T_LEN * DIM * 2;   // T*3072*2
static const size_t OFF_ATTN = OFF_QKV  + (size_t)T_LEN * QKV_N * 2; // T*1024*2

// ------------------------- WMMA fragment loaders ---------------------------
// A fragment (16x32 f16, ISA 7.12.2): lanes 0-15 row M=l hold K {k0..k0+7, k0+16..k0+23};
// lanes 16-31 row M=l hold K {k0+8..k0+15, k0+24..k0+31}.
__device__ __forceinline__ v16h load_frag_a(const h16* sm, int ld, int m0, int k0, int lane) {
    const int half = lane >> 4;
    const int l    = lane & 15;
    const h16* p = sm + (size_t)(m0 + l) * ld + k0 + half * 8;
    union { v16h v; uint4 q[2]; } u;
    u.q[0] = *(const uint4*)(p);
    u.q[1] = *(const uint4*)(p + 16);
    return u.v;
}

// B fragment (32x16 f16): lane = output column, lanes 0-15 hold K {k0..k0+7, k0+16..k0+23},
// lanes 16-31 hold K {k0+8..k0+15, k0+24..k0+31}. Source is an (N x K) row-major tile, i.e.
// weight rows load directly as B columns.
__device__ __forceinline__ v16h load_frag_b(const h16* sm, int ld, int n0, int k0, int lane) {
    const int half = lane >> 4;
    const int l    = lane & 15;
    const h16* p = sm + (size_t)(n0 + l) * ld + k0 + half * 8;
    union { v16h v; uint4 q[2]; } u;
    u.q[0] = *(const uint4*)(p);
    u.q[1] = *(const uint4*)(p + 16);
    return u.v;
}

#if USE_ASYNC_LDS
// The builtin expects pointers to a 16-byte GCC int vector (per hipcc
// diagnostic): global side in AS1 (__device__), LDS side in AS3.
typedef int v4i_g __attribute__((vector_size(16)));
typedef __attribute__((address_space(1))) v4i_g as1_v4i;
typedef __attribute__((address_space(3))) v4i_g as3_v4i;
// Copy 32 contiguous bytes (2 x b128) global -> LDS on the ASYNCcnt path.
// The immediate offset applies to both the global and LDS addresses (ISA
// GLOBAL_LOAD_ASYNC_TO_LDS semantics), and our tile layout keeps both +16.
__device__ __forceinline__ void async_copy32(const void* g, void* l) {
    as1_v4i* gp = (as1_v4i*)(uintptr_t)g;
    as3_v4i* lp = (as3_v4i*)(uint32_t)(uintptr_t)l;   // low 32 bits = LDS offset
    __builtin_amdgcn_global_load_async_to_lds_b128(gp, lp, 0, 0);
    __builtin_amdgcn_global_load_async_to_lds_b128(gp, lp, 16, 0);
}
#endif

// ------------------------- kernel 0: f32 -> f16 cast -----------------------
__global__ void cvt_f32_to_f16(const float* __restrict__ in, h16* __restrict__ out, int n) {
    int i = blockIdx.x * blockDim.x + threadIdx.x;
    if (i < n) out[i] = (h16)in[i];
}

// ------------------------- kernel 1: LayerNorm -----------------------------
// One block per time step t; x is (C, T) so column t is strided by T.
__global__ __launch_bounds__(256) void layernorm_kernel(
    const float* __restrict__ x, const float* __restrict__ gamma,
    const float* __restrict__ beta, h16* __restrict__ normed) {
    const int t   = blockIdx.x;
    const int tid = threadIdx.x;
    float v[4];
    float s = 0.f, s2 = 0.f;
#pragma unroll
    for (int i = 0; i < 4; ++i) {
        int   c   = tid + i * 256;
        float val = x[(size_t)c * T_LEN + t];
        v[i] = val; s += val; s2 += val * val;
    }
    __shared__ float red0[256];
    __shared__ float red1[256];
    red0[tid] = s; red1[tid] = s2;
    __syncthreads();
    for (int off = 128; off > 0; off >>= 1) {
        if (tid < off) { red0[tid] += red0[tid + off]; red1[tid] += red1[tid + off]; }
        __syncthreads();
    }
    const float mu  = red0[0] * (1.0f / DIM);
    const float var = red1[0] * (1.0f / DIM) - mu * mu;
    const float rs  = rsqrtf(var + 1e-5f);
#pragma unroll
    for (int i = 0; i < 4; ++i) {
        int c = tid + i * 256;
        normed[(size_t)t * DIM + c] = (h16)((v[i] - mu) * rs * gamma[c] + beta[c]);
    }
}

// ------------------------- WMMA block GEMM core ----------------------------
// C[M,N] += A[M,K] * B[N,K]^T.  Block tile 128x128, 8 waves as 4(M) x 2(N),
// wave tile 32x64 = 2x4 accumulators.  K staged in 32-wide LDS tiles.
#define GE_BM 128
#define GE_BN 128
#define GE_BK 32
#define LDS_P (GE_BK + 8)   // 40 halfs = 80 B row stride: 16B-aligned, conflict-free

__device__ __forceinline__ void wmma_tile(const h16* cA, const h16* cB,
                                          int wm0, int wn0, int lane, v8f acc[2][4]) {
#pragma unroll
    for (int i = 0; i < 2; ++i) {
        const v16h af = load_frag_a(cA, LDS_P, wm0 + i * 16, 0, lane);
#pragma unroll
        for (int j = 0; j < 4; ++j) {
            const v16h bf = load_frag_b(cB, LDS_P, wn0 + j * 16, 0, lane);
            acc[i][j] = __builtin_amdgcn_wmma_f32_16x16x32_f16(
                false, af, false, bf, (short)0, acc[i][j], false, false);
        }
    }
}

__device__ __forceinline__ void gemm_block_f16(
    const h16* __restrict__ A, const h16* __restrict__ B,
    int K, int m0g, int n0g, v8f acc[2][4]) {
    const int tid  = threadIdx.x;
    const int lane = tid & 31;
    const int wave = tid >> 5;      // 0..7
    const int wm0  = (wave >> 1) * 32;
    const int wn0  = (wave & 1) * 64;

    const v8f zero = {0.f, 0.f, 0.f, 0.f, 0.f, 0.f, 0.f, 0.f};
#pragma unroll
    for (int i = 0; i < 2; ++i)
#pragma unroll
        for (int j = 0; j < 4; ++j) acc[i][j] = zero;

    const int ldrow = tid >> 1;        // 0..127
    const int ldcol = (tid & 1) * 16;  // 0 or 16 (halfs)
    const h16* gA = A + (size_t)(m0g + ldrow) * K + ldcol;
    const h16* gB = B + (size_t)(n0g + ldrow) * K + ldcol;

#if USE_ASYNC_LDS
    // Double-buffered ASYNCcnt pipeline: tile k+1 streams into buf^1 while
    // WMMAs consume buf.  4 async ops/tile/wave; waits are in-order so
    // s_wait_asynccnt 4 releases exactly the previous tile.
    __shared__ h16 sA[2][GE_BM * LDS_P];
    __shared__ h16 sB[2][GE_BN * LDS_P];
    h16* lA = &sA[0][ldrow * LDS_P + ldcol];
    h16* lB = &sB[0][ldrow * LDS_P + ldcol];
    const size_t bufstep = (size_t)GE_BM * LDS_P;  // halfs per buffer

    async_copy32(gA, lA);
    async_copy32(gB, lB);
    int buf = 0;
    for (int k0 = 0; k0 < K; k0 += GE_BK) {
        const bool more = (k0 + GE_BK) < K;
        if (more) {
            async_copy32(gA + k0 + GE_BK, lA + (buf ^ 1) * bufstep);
            async_copy32(gB + k0 + GE_BK, lB + (buf ^ 1) * bufstep);
            __builtin_amdgcn_s_wait_asynccnt(4);
        } else {
            __builtin_amdgcn_s_wait_asynccnt(0);
        }
        __syncthreads();
        wmma_tile(sA[buf], sB[buf], wm0, wn0, lane, acc);
        __syncthreads();   // all reads of buf done before it is refilled
        buf ^= 1;
    }
#else
    __shared__ h16 sA[GE_BM * LDS_P];
    __shared__ h16 sB[GE_BN * LDS_P];
    for (int k0 = 0; k0 < K; k0 += GE_BK) {
        const uint4* ga = (const uint4*)(gA + k0);
        const uint4  a0 = ga[0], a1 = ga[1];
        const uint4* gb = (const uint4*)(gB + k0);
        const uint4  b0 = gb[0], b1 = gb[1];
        __syncthreads();  // previous iteration's fragment reads done
        *(uint4*)(sA + ldrow * LDS_P + ldcol)     = a0;
        *(uint4*)(sA + ldrow * LDS_P + ldcol + 8) = a1;
        *(uint4*)(sB + ldrow * LDS_P + ldcol)     = b0;
        *(uint4*)(sB + ldrow * LDS_P + ldcol + 8) = b1;
        __syncthreads();
        wmma_tile(sA, sB, wm0, wn0, lane, acc);
    }
#endif
}

// ------------------------- kernel 2: QKV GEMM ------------------------------
__global__ __launch_bounds__(256) void qkv_gemm_kernel(
    const h16* __restrict__ A /* normed T x DIM */,
    const h16* __restrict__ B /* w_qkv 3072 x 1024 */,
    h16* __restrict__ C       /* qkv T x 3072 */) {
    const int m0g = blockIdx.y * GE_BM;
    const int n0g = blockIdx.x * GE_BN;
    v8f acc[2][4];
    gemm_block_f16(A, B, DIM, m0g, n0g, acc);

    const int lane = threadIdx.x & 31;
    const int wave = threadIdx.x >> 5;
    const int wm0  = (wave >> 1) * 32;
    const int wn0  = (wave & 1) * 64;
    const int l    = lane & 15;
    const int half = lane >> 4;
#pragma unroll
    for (int i = 0; i < 2; ++i)
#pragma unroll
        for (int j = 0; j < 4; ++j)
#pragma unroll
            for (int r = 0; r < 8; ++r) {
                const int row = m0g + wm0 + i * 16 + (half ? r + 8 : r);
                const int col = n0g + wn0 + j * 16 + l;
                C[(size_t)row * QKV_N + col] = (h16)acc[i][j][r];
            }
}

// ------------------------- kernel 3: windowed attention --------------------
// One block per (window, head); 64 threads = 2 waves.  RoPE fused into the
// LDS fill; QK^T and PV both on WMMA.
__global__ __launch_bounds__(64) void attn_kernel(
    const h16* __restrict__ qkv, h16* __restrict__ attn_out) {
    const int w   = blockIdx.x;   // 0..511
    const int h   = blockIdx.y;   // 0..15
    const int t0  = w * WINDOW;
    const int tid = threadIdx.x;
    const int lane = tid & 31;
    const int wv   = tid >> 5;    // 0 or 1

    __shared__ h16   sQ[WINDOW * 72];   // 32x64, row stride 72 (144 B, 16B-aligned)
    __shared__ h16   sK[WINDOW * 72];
    __shared__ h16   sVt[DHEAD * 40];   // V transposed: 64x32, stride 40
    __shared__ float sS[WINDOW * 33];   // logits
    __shared__ h16   sP[WINDOW * 40];   // softmax probs, stride 40

    const float LOG1E4_OVER32 = 0.28782313662425572f;  // ln(10000)/32

    for (int i = tid; i < WINDOW * DHEAD; i += 64) {
        const int n = i >> 6;       // window position
        const int d = i & 63;       // head dim
        const int j = d & 31;       // rotary frequency index
        const float f  = (float)n * __expf(-(float)j * LOG1E4_OVER32);
        const float cf = __cosf(f), sf = __sinf(f);
        const size_t base = (size_t)(t0 + n) * QKV_N + h * DHEAD;
        const float qv = (float)qkv[base + d];
        const float kv = (float)qkv[base + DIM + d];
        const int   dp = (d < 32) ? d + 32 : d - 32;
        const float qp = (float)qkv[base + dp];
        const float kp = (float)qkv[base + DIM + dp];
        const float sg = (d < 32) ? -1.f : 1.f;
        sQ[n * 72 + d]  = (h16)(qv * cf + sg * qp * sf);
        sK[n * 72 + d]  = (h16)(kv * cf + sg * kp * sf);
        sVt[d * 40 + n] = qkv[base + 2 * DIM + d];
    }
    __syncthreads();

    // S = Q K^T : wave wv computes rows [wv*16, wv*16+16)
    const int m0 = wv * 16;
    v8f sacc[2];
#pragma unroll
    for (int nf = 0; nf < 2; ++nf) {
        v8f a = {0.f, 0.f, 0.f, 0.f, 0.f, 0.f, 0.f, 0.f};
#pragma unroll
        for (int k0 = 0; k0 < DHEAD; k0 += 32) {
            const v16h qa = load_frag_a(sQ, 72, m0, k0, lane);
            const v16h kb = load_frag_b(sK, 72, nf * 16, k0, lane);
            a = __builtin_amdgcn_wmma_f32_16x16x32_f16(
                false, qa, false, kb, (short)0, a, false, false);
        }
        sacc[nf] = a;
    }
    {
        const int l = lane & 15, half = lane >> 4;
#pragma unroll
        for (int nf = 0; nf < 2; ++nf)
#pragma unroll
            for (int r = 0; r < 8; ++r) {
                const int row = m0 + (half ? r + 8 : r);
                const int col = nf * 16 + l;
                sS[row * 33 + col] = sacc[nf][r];
            }
    }
    __syncthreads();

    // softmax over each row of 32 logits (scale = 1/sqrt(64))
    if (tid < WINDOW) {
        const float scale = 0.125f;
        float vals[32];
        float mx = -3.4e38f;
#pragma unroll
        for (int j = 0; j < 32; ++j) {
            vals[j] = sS[tid * 33 + j] * scale;
            mx = fmaxf(mx, vals[j]);
        }
        float sum = 0.f;
#pragma unroll
        for (int j = 0; j < 32; ++j) { vals[j] = __expf(vals[j] - mx); sum += vals[j]; }
        const float inv = 1.f / sum;
#pragma unroll
        for (int j = 0; j < 32; ++j) sP[tid * 40 + j] = (h16)(vals[j] * inv);
    }
    __syncthreads();

    // O = P V : wave wv does rows m0..m0+15 across DHEAD (4 N-frags, K=32)
#pragma unroll
    for (int nf = 0; nf < 4; ++nf) {
        v8f a = {0.f, 0.f, 0.f, 0.f, 0.f, 0.f, 0.f, 0.f};
        const v16h pa = load_frag_a(sP, 40, m0, 0, lane);
        const v16h vb = load_frag_b(sVt, 40, nf * 16, 0, lane);
        a = __builtin_amdgcn_wmma_f32_16x16x32_f16(
            false, pa, false, vb, (short)0, a, false, false);
        const int l = lane & 15, half = lane >> 4;
#pragma unroll
        for (int r = 0; r < 8; ++r) {
            const int row = t0 + m0 + (half ? r + 8 : r);
            const int col = h * DHEAD + nf * 16 + l;
            attn_out[(size_t)row * DIM + col] = (h16)a[r];
        }
    }
}

// ------------------------- kernel 4: out-proj + residual -------------------
// out[c, t] = (attn @ w_out^T)[t, c] + x[c, t]   (f32 output, transposed)
__global__ __launch_bounds__(256) void outproj_kernel(
    const h16* __restrict__ A /* attn T x 1024 */,
    const h16* __restrict__ B /* w_out 1024 x 1024 */,
    const float* __restrict__ x, float* __restrict__ out) {
    const int m0g = blockIdx.y * GE_BM;
    const int n0g = blockIdx.x * GE_BN;
    v8f acc[2][4];
    gemm_block_f16(A, B, DIM, m0g, n0g, acc);

    const int lane = threadIdx.x & 31;
    const int wave = threadIdx.x >> 5;
    const int wm0  = (wave >> 1) * 32;
    const int wn0  = (wave & 1) * 64;
    const int l    = lane & 15;
    const int half = lane >> 4;
#pragma unroll
    for (int i = 0; i < 2; ++i)
#pragma unroll
        for (int j = 0; j < 4; ++j)
#pragma unroll
            for (int r = 0; r < 8; ++r) {
                const int row = m0g + wm0 + i * 16 + (half ? r + 8 : r);  // t
                const int col = n0g + wn0 + j * 16 + l;                   // c
                const size_t idx = (size_t)col * T_LEN + row;
                out[idx] = acc[i][j][r] + x[idx];
            }
}

// ------------------------- launcher ----------------------------------------
extern "C" void kernel_launch(void* const* d_in, const int* in_sizes, int n_in,
                              void* d_out, int out_size, void* d_ws, size_t ws_size,
                              hipStream_t stream) {
    const float* x     = (const float*)d_in[0];
    const float* w_qkv = (const float*)d_in[1];
    const float* w_out = (const float*)d_in[2];
    const float* gamma = (const float*)d_in[3];
    const float* beta  = (const float*)d_in[4];
    float* out = (float*)d_out;

    char* ws = (char*)d_ws;
    h16* wqkv_h = (h16*)(ws + OFF_WQKV);
    h16* wout_h = (h16*)(ws + OFF_WOUT);
    h16* normed = (h16*)(ws + OFF_NORM);
    h16* qkv    = (h16*)(ws + OFF_QKV);
    h16* attn   = (h16*)(ws + OFF_ATTN);

    {
        int n = QKV_N * DIM;
        cvt_f32_to_f16<<<(n + 255) / 256, 256, 0, stream>>>(w_qkv, wqkv_h, n);
        n = DIM * DIM;
        cvt_f32_to_f16<<<(n + 255) / 256, 256, 0, stream>>>(w_out, wout_h, n);
    }
    layernorm_kernel<<<T_LEN, 256, 0, stream>>>(x, gamma, beta, normed);
    qkv_gemm_kernel<<<dim3(QKV_N / GE_BN, T_LEN / GE_BM), 256, 0, stream>>>(normed, wqkv_h, qkv);
    attn_kernel<<<dim3(NWIN, HEADS), 64, 0, stream>>>(qkv, attn);
    outproj_kernel<<<dim3(DIM / GE_BN, T_LEN / GE_BM), 256, 0, stream>>>(attn, wout_h, x, out);
}
